// KANLayer_90787018702997
// MI455X (gfx1250) — matrix operand order repaired and checked by reading, not compiled, not run on hardware
//
#include <hip/hip_runtime.h>
#include <stdint.h>

typedef __attribute__((ext_vector_type(16))) _Float16 v16h;
typedef __attribute__((ext_vector_type(8)))  _Float16 v8h;
typedef __attribute__((ext_vector_type(8)))  float    v8f;

// exact parameter types for __builtin_amdgcn_global_load_async_to_lds_b128:
// (global int4*, local int4*, imm offset, imm cpol)
typedef int v4i __attribute__((vector_size(4 * sizeof(int))));
typedef __attribute__((address_space(1))) v4i* as1_v4i_ptr;
typedef __attribute__((address_space(3))) v4i* as3_v4i_ptr;

#define IN_DIM   128
#define OUT_DIM  128
#define NB       18                   // N_GRID + ORDER - 1
#define BATCH    8192
#define KDIM     (IN_DIM * NB)        // 2304 = 72 * 32
#define MTILE    128                  // batch rows per block
#define CH_FEATS 16                   // in-features per K-chunk
#define CH_K     (CH_FEATS * NB)      // 288 halves = 9 WMMA K-steps
#define CH_STEPS (CH_K / 32)          // 9
#define NCHUNK   (IN_DIM / CH_FEATS)  // 8

#define ALDS_HALVES (MTILE * CH_K)      // 36864 halves = 73728 B
#define WLDS_HALVES (CH_STEPS * 4096)   // 36864 halves = 73728 B
#define SMEM_BYTES  (ALDS_HALVES * 2 + WLDS_HALVES * 2 + MTILE * 4)  // 147968 B

#if defined(__has_builtin)
#if __has_builtin(__builtin_amdgcn_global_load_async_to_lds_b128) && \
    __has_builtin(__builtin_amdgcn_s_wait_asynccnt)
#define HAVE_ASYNC_LDS 1
#endif
#endif

// ---------------------------------------------------------------------------
// Kernel 1: convert coef (f32, [out][in][18]) into f16 laid out exactly as
// WMMA B fragments per 32x16 tile:
//   element = ks*4096 + nt*512 + lane*16 + hh
//   k = ks*32 + (lane>>4)*16 + hh ; n = nt*16 + (lane&15) ; k = i*18 + t
// ---------------------------------------------------------------------------
__global__ void kan_wswz_kernel(const float* __restrict__ coef,
                                _Float16* __restrict__ W) {
  const int tid  = blockIdx.x * blockDim.x + threadIdx.x;
  const int hh   = tid & 15;
  const int lane = (tid >> 4) & 31;
  const int nt   = (tid >> 9) & 7;
  const int ks   = tid >> 12;

  const int n    = nt * 16 + (lane & 15);
  const int krel = (lane >> 4) * 16 + hh;
  const int k    = ks * 32 + krel;
  const int i    = k / NB;
  const int t    = k - i * NB;

  W[tid] = (_Float16)coef[((size_t)n * IN_DIM + i) * NB + t];
}

// ---------------------------------------------------------------------------
// Kernel 2 (fused): per block of 128 batch rows, loop over 8 K-chunks:
//  phase A: async-DMA the W chunk into LDS while computing tanh + cubic
//           B-spline basis tile into LDS (ds_add_f32 row-mean accumulation),
//  phase B: 9 K-steps x 8 N-tiles of v_wmma_f32_16x16x32_f16 from LDS.
// ---------------------------------------------------------------------------
__global__ void __launch_bounds__(256) kan_fused_kernel(
    const float* __restrict__ x,
    const _Float16* __restrict__ W,
    float* __restrict__ out) {
  extern __shared__ char smem[];
  _Float16* Alds = (_Float16*)smem;                               // 73728 B
  _Float16* Wlds = (_Float16*)(smem + ALDS_HALVES * 2);           // 73728 B
  float*    srow = (float*)(smem + ALDS_HALVES * 2 + WLDS_HALVES * 2);

  const int tid     = threadIdx.x;
  const int lane    = tid & 31;
  const int wave    = tid >> 5;
  const int group   = lane >> 4;
  const int lane16  = lane & 15;
  const int rowbase = blockIdx.x * MTILE;

  if (tid < MTILE) srow[tid] = 0.0f;

  v8f c[8];
#pragma unroll
  for (int nt = 0; nt < 8; ++nt) {
#pragma unroll
    for (int e = 0; e < 8; ++e) c[nt][e] = 0.0f;
  }

  const float h  = 2.0f / 15.0f;       // knot spacing
  const float g0 = -1.0f - 3.0f * h;   // extended grid start (order 3)

  for (int c0 = 0; c0 < NCHUNK; ++c0) {
    __syncthreads();  // previous chunk's LDS reads are done

    // ---- stage W chunk (global -> LDS), overlapped with basis compute ----
    {
      const uint4* src = (const uint4*)(W + (size_t)c0 * WLDS_HALVES);
#if HAVE_ASYNC_LDS
#pragma unroll
      for (int q = 0; q < 18; ++q) {  // 18 * 256 lanes * 16B = 73728 B
        const int idx = tid + 256 * q;
        __builtin_amdgcn_global_load_async_to_lds_b128(
            (as1_v4i_ptr)(uintptr_t)(src + idx),
            (as3_v4i_ptr)(uint32_t)(uintptr_t)((uint4*)Wlds + idx),
            0, 0);
      }
#else
      uint4* dst = (uint4*)Wlds;
#pragma unroll
      for (int q = 0; q < 18; ++q) {
        const int idx = tid + 256 * q;
        dst[idx] = src[idx];
      }
#endif
    }

    // ---- basis tile: 128 rows x 16 features -> LDS (f16) ----
    for (int q = 0; q < 8; ++q) {
      const int b  = 16 * q + (tid >> 4);  // local row 0..127
      const int il = tid & 15;             // feature within chunk
      const int i  = c0 * CH_FEATS + il;

      const float xt = tanhf(x[(size_t)(rowbase + b) * IN_DIM + i]);
      atomicAdd(&srow[b], xt);             // ds_add_f32 row-mean accumulation

      float B[21];
#pragma unroll
      for (int j = 0; j < 21; ++j) {
        const float gj = g0 + (float)j * h;
        B[j] = (xt >= gj && xt < gj + h) ? 1.0f : 0.0f;  // degree 0
      }
#pragma unroll
      for (int k = 1; k <= 3; ++k) {
        const float inv = 1.0f / ((float)k * h);
#pragma unroll
        for (int j = 0; j <= 20 - k; ++j) {
          const float gj    = g0 + (float)j * h;
          const float left  = (xt - gj) * inv;
          const float right = (gj + (float)(k + 1) * h - xt) * inv;
          B[j] = left * B[j] + right * B[j + 1];
        }
      }
      _Float16* dst = Alds + b * CH_K + il * NB;
#pragma unroll
      for (int t = 0; t < NB; ++t) dst[t] = (_Float16)B[t];
    }

#if HAVE_ASYNC_LDS
    __builtin_amdgcn_s_wait_asynccnt(0);
#endif
    __syncthreads();

    // ---- WMMA: 9 K-steps x 8 N-tiles, A/B fragments from LDS ----
    const _Float16* arow = Alds + (wave * 16 + lane16) * CH_K;
    for (int s = 0; s < CH_STEPS; ++s) {
      const int kk = s * 32;
      // A fragment: lane-group 0 holds K {kk+0..15},
      //             lane-group 1 holds K {kk+16..31}
      v8h a0 = *(const v8h*)(arow + kk + group * 16);
      v8h a1 = *(const v8h*)(arow + kk + group * 16 + 8);
      v16h a;
#pragma unroll
      for (int e = 0; e < 8; ++e) { a[e] = a0[e]; a[e + 8] = a1[e]; }

      const _Float16* wb = Wlds + s * 4096 + (size_t)lane * 16;
#pragma unroll
      for (int nt = 0; nt < 8; ++nt) {
        v16h bf = *(const v16h*)(wb + nt * 512);
        c[nt] = __builtin_amdgcn_wmma_f32_16x16x32_f16(
            /*neg_a=*/false, a, /*neg_b=*/false, bf,
            /*c_mod=*/(short)0, c[nt], /*reuse_a=*/false, /*reuse_b=*/false);
      }
    }
  }

  __syncthreads();

  // ---- epilogue: out = (sum_i xt + spline_acc) / 128 ----
  // D layout: lane 0-15 -> N=lane, VGPR r -> M=r; lanes 16-31 -> M=r+8.
  float m[8];
#pragma unroll
  for (int r = 0; r < 8; ++r) m[r] = srow[wave * 16 + r + 8 * group];

#pragma unroll
  for (int nt = 0; nt < 8; ++nt) {
    const int o = nt * 16 + lane16;
#pragma unroll
    for (int r = 0; r < 8; ++r) {
      const int b = rowbase + wave * 16 + r + 8 * group;
      out[(size_t)b * OUT_DIM + o] = (m[r] + c[nt][r]) * (1.0f / IN_DIM);
    }
  }
}

// ---------------------------------------------------------------------------
extern "C" void kernel_launch(void* const* d_in, const int* in_sizes, int n_in,
                              void* d_out, int out_size, void* d_ws, size_t ws_size,
                              hipStream_t stream) {
  const float* x    = (const float*)d_in[0];   // [8192,128] f32
  const float* coef = (const float*)d_in[1];   // [16384,18] f32
  float* out        = (float*)d_out;           // [8192,128] f32

  _Float16* Wswz = (_Float16*)d_ws;            // 589,824 B

  kan_wswz_kernel<<<(KDIM * OUT_DIM) / 256, 256, 0, stream>>>(coef, Wswz);

  (void)hipFuncSetAttribute((const void*)kan_fused_kernel,
                            hipFuncAttributeMaxDynamicSharedMemorySize,
                            SMEM_BYTES);
  kan_fused_kernel<<<BATCH / MTILE, 256, SMEM_BYTES, stream>>>(x, Wswz, out);
}
